// Generator_11433202942753
// MI455X (gfx1250) — matrix operand order repaired and checked by reading, not compiled, not run on hardware
//
#include <hip/hip_runtime.h>
#include <math.h>

// ---------------------------------------------------------------------------
// CDNA5 (gfx1250, wave32) WMMA types & wrapper
// ---------------------------------------------------------------------------
typedef __attribute__((ext_vector_type(16))) _Float16 v16h;
typedef __attribute__((ext_vector_type(8)))  _Float16 v8h;
typedef __attribute__((ext_vector_type(8)))  float    v8f;

__device__ __forceinline__ v8f wmma16x16x32(v16h a, v16h b, v8f c) {
    // D = A(16x32 f16) * B(32x16 f16) + C(16x16 f32)
    return __builtin_amdgcn_wmma_f32_16x16x32_f16(
        false, a, false, b, (short)0, c, false, false);
}

__device__ __forceinline__ float gelu_exact(float x) {
    return 0.5f * x * (1.0f + erff(x * 0.70710678118654752440f));
}

// hardware sin/cos take input in revolutions: v_sin_f32(x) = sin(2*pi*x)
__device__ __forceinline__ float hw_cospi2(float rev) {
    return __builtin_amdgcn_cosf(rev);
}
__device__ __forceinline__ float hw_sinpi2(float rev) {
    return __builtin_amdgcn_sinf(rev);
}

// ---------------------------------------------------------------------------
// fc0: [B,6400,8] + label[B,4] --> h[B,64,6400] (channel-major), exact GELU
// ---------------------------------------------------------------------------
__global__ void __launch_bounds__(256)
fc0_kernel(const float* __restrict__ x, const float* __restrict__ lab,
           const float* __restrict__ W, const float* __restrict__ bias,
           float* __restrict__ h) {
    __shared__ float sW[12 * 64];
    __shared__ float sB[64];
    int tid = threadIdx.x;
    for (int i = tid; i < 12 * 64; i += 256) sW[i] = W[i];
    if (tid < 64) sB[tid] = bias[tid];
    __syncthreads();
    int idx = blockIdx.x * 256 + tid;          // over B*6400
    int s = idx % 6400;
    int b = idx / 6400;
    float in[12];
#pragma unroll
    for (int k = 0; k < 8; ++k) in[k] = x[((size_t)b * 6400 + s) * 8 + k];
#pragma unroll
    for (int k = 0; k < 4; ++k) in[8 + k] = lab[b * 4 + k];
    for (int c = 0; c < 64; ++c) {
        float acc = sB[c];
#pragma unroll
        for (int k = 0; k < 12; ++k) acc += in[k] * sW[k * 64 + c];
        h[((size_t)b * 64 + c) * 6400 + s] = gelu_exact(acc);
    }
}

// ---------------------------------------------------------------------------
// Forward DFT as implicit-basis GEMM (WMMA):
//   spec[row][mode] = sum_n inp[row][n] * exp(-2*pi*i*n*mode/L), mode<modes
// Basis tile (32 x 64 complex, f16) generated on the fly into LDS.
// Phase (n*mode mod L) maintained as an arithmetic progression mod L:
// no division in the loop; hardware sin/cos consume revolutions directly.
// block = 256 threads (8 waves); macro tile: M=128 rows, N=64 modes.
// ---------------------------------------------------------------------------
__global__ void __launch_bounds__(256)
dft_fwd_kernel(const float* __restrict__ inp, float* __restrict__ spec,
               int R, int L, int modes) {
    __shared__ _Float16 bcos[4 * 32 * 16];
    __shared__ _Float16 bsin[4 * 32 * 16];
    int tid   = threadIdx.x;
    int lane  = tid & 31;
    int wave  = tid >> 5;
    int nb0   = blockIdx.x * 64;                 // mode base
    int rowBase = blockIdx.y * 128 + wave * 16;  // M base for this wave
    int mrow  = lane & 15;
    int khalf = lane >> 4;                       // A-frag K split
    int row   = rowBase + mrow;
    const float* rowPtr = inp + (size_t)((row < R) ? row : 0) * L;

    // --- per-thread basis-generation state (thread owns one mode, 8 krel)
    int nrel   = tid & 63;
    int md     = nb0 + nrel;                     // may exceed modes; harmless
    int krbase = (tid >> 6) * 8;                 // 0,8,16,24
    int step32 = (32 * md) % L;                  // phase advance per K-step
    int phase  = (krbase * md) % L;              // (kb+krbase)*md mod L, kb=0
    int emd[8];
#pragma unroll
    for (int e = 0; e < 8; ++e) emd[e] = (e * md) % L;
    int tt    = nrel >> 4;
    int dl    = (nrel & 15) + 16 * ((krbase >> 3) & 1);
    int lbase = ((tt * 32 + dl) << 4) + ((krbase & 16) >> 1);

    v8f accRe[4] = {};
    v8f accIm[4] = {};
    float invL = 1.0f / (float)L;

    for (int kb = 0; kb < L; kb += 32) {
        // cooperative basis generation: 8 consecutive halfs -> 1 b128 store
        v8h c8, s8;
#pragma unroll
        for (int e = 0; e < 8; ++e) {
            int ph = phase + emd[e];
            if (ph >= L) ph -= L;
            float fr = (float)ph * invL;         // revolutions in [0,1)
            c8[e] = (_Float16)hw_cospi2(fr);     // cos(-t) =  cos(t)
            s8[e] = (_Float16)(-hw_sinpi2(fr));  // sin(-t) = -sin(t)
        }
        *(v8h*)&bcos[lbase] = c8;
        *(v8h*)&bsin[lbase] = s8;
        phase += step32;
        if (phase >= L) phase -= L;
        __syncthreads();

        // A fragment (16x32 f16): lanes 0-15 hold K 0..7 & 16..23
        v16h af;
        {
            int ks = khalf * 8;
#pragma unroll
            for (int j = 0; j < 8; ++j) {
                af[j]     = (_Float16)rowPtr[kb + ks + j];
                af[8 + j] = (_Float16)rowPtr[kb + 16 + ks + j];
            }
        }
#pragma unroll
        for (int t = 0; t < 4; ++t) {
            const v16h* pc = (const v16h*)&bcos[(t * 32 + lane) << 4];
            const v16h* ps = (const v16h*)&bsin[(t * 32 + lane) << 4];
            accRe[t] = wmma16x16x32(af, *pc, accRe[t]);
            accIm[t] = wmma16x16x32(af, *ps, accIm[t]);
        }
        __syncthreads();
    }
#pragma unroll
    for (int v = 0; v < 8; ++v) {
        int orow = rowBase + v + 8 * khalf;
        if (orow < R) {
#pragma unroll
            for (int t = 0; t < 4; ++t) {
                int mdo = nb0 + t * 16 + (lane & 15);
                if (mdo < modes) {
                    float2 val = make_float2(accRe[t][v], accIm[t][v]);
                    ((float2*)spec)[(size_t)orow * modes + mdo] = val;
                }
            }
        }
    }
}

// ---------------------------------------------------------------------------
// Per-mode complex einsum (WMMA): Y[b,co,k] = sum_ci X[b,ci,k] * W[ci,co,k]
// irfft weights folded into output: scale = (k==0?1:2)/Lout.
// block = 256 (8 waves): wave = (mtile in {0,1}) x (ntile in 0..3)
// ---------------------------------------------------------------------------
__global__ void __launch_bounds__(256)
spec_einsum_kernel(const float* __restrict__ X, const float* __restrict__ W,
                   float* __restrict__ Y, int Ci, int Co, int modes,
                   float invLout) {
    int k      = blockIdx.x;
    int coBase = blockIdx.y * 64;
    int lane   = threadIdx.x & 31;
    int wave   = threadIdx.x >> 5;
    int mtile  = wave & 1;
    int ntile  = wave >> 1;
    int mrow   = lane & 15;
    int khalf  = lane >> 4;
    int b      = mtile * 16 + mrow;
    int coW    = coBase + ntile * 16 + mrow;
    if (coW >= Co) coW = Co - 1;   // clamp loads; writes are guarded

    const float2* Xc = (const float2*)X;
    const float2* Wc = (const float2*)W;

    v8f accRe = {};
    v8f accIm = {};

    for (int cb = 0; cb < Ci; cb += 32) {
        v16h Ar, Ai;
        {
            int ks = khalf * 8;
#pragma unroll
            for (int j = 0; j < 8; ++j) {
                int ci0 = cb + ks + j;
                int ci1 = ci0 + 16;
                float2 x0 = Xc[((size_t)b * Ci + ci0) * modes + k];
                float2 x1 = Xc[((size_t)b * Ci + ci1) * modes + k];
                Ar[j]     = (_Float16)x0.x;  Ai[j]     = (_Float16)x0.y;
                Ar[8 + j] = (_Float16)x1.x;  Ai[8 + j] = (_Float16)x1.y;
            }
        }
        v16h Wr, Wi, nWi;
#pragma unroll
        for (int j = 0; j < 16; ++j) {
            int ci = cb + 8 * khalf + (j & 7) + 16 * (j >> 3);
            float2 w = Wc[((size_t)ci * Co + coW) * modes + k];
            Wr[j]  = (_Float16)w.x;
            Wi[j]  = (_Float16)w.y;
            nWi[j] = (_Float16)(-w.y);
        }
        accRe = wmma16x16x32(Ar, Wr,  accRe);
        accRe = wmma16x16x32(Ai, nWi, accRe);
        accIm = wmma16x16x32(Ar, Wi,  accIm);
        accIm = wmma16x16x32(Ai, Wr,  accIm);
    }

    float scale = ((k == 0) ? 1.0f : 2.0f) * invLout;
    float2* Yc = (float2*)Y;
#pragma unroll
    for (int v = 0; v < 8; ++v) {
        int ob = mtile * 16 + v + 8 * khalf;
        int oc = coBase + ntile * 16 + (lane & 15);
        if (oc < Co) {
            float2 val = make_float2(accRe[v] * scale, accIm[v] * scale);
            Yc[((size_t)ob * Co + oc) * modes + k] = val;
        }
    }
}

// ---------------------------------------------------------------------------
// Inverse DFT as implicit-basis GEMM (WMMA):
//   out[row][s] = sum_{t<2m} A[row][t] * basis(t,s)
//   basis(2k,s)=cos(2*pi*k*s/L), basis(2k+1,s)=-sin(2*pi*k*s/L)
//   (per-mode 1/L and x2 weights already folded into A by the einsum)
// Same division-free phase recurrence as the forward kernel.
// ---------------------------------------------------------------------------
__global__ void __launch_bounds__(256)
dft_inv_kernel(const float* __restrict__ A, float* __restrict__ out,
               int R, int K2, int L) {
    __shared__ _Float16 bb[4 * 32 * 16];
    int tid   = threadIdx.x;
    int lane  = tid & 31;
    int wave  = tid >> 5;
    int nb0   = blockIdx.x * 64;                 // output position base
    int rowBase = blockIdx.y * 128 + wave * 16;
    int mrow  = lane & 15;
    int khalf = lane >> 4;
    int row   = rowBase + mrow;
    const float* rowPtr = A + (size_t)((row < R) ? row : 0) * K2;

    // --- per-thread basis state: thread owns one output position s
    int nrel   = tid & 63;
    int s      = nb0 + nrel;
    int krbase = (tid >> 6) * 8;                 // even
    int step16 = (16 * s) % L;                   // kk advances 16 per K-step
    int phase  = ((krbase >> 1) * s) % L;        // kk_base*s mod L at kb=0
    int esm[4];
#pragma unroll
    for (int q = 0; q < 4; ++q) esm[q] = (q * s) % L;
    int tt    = nrel >> 4;
    int dl    = (nrel & 15) + 16 * ((krbase >> 3) & 1);
    int lbase = ((tt * 32 + dl) << 4) + ((krbase & 16) >> 1);

    v8f acc[4] = {};
    float invL = 1.0f / (float)L;

    for (int kb = 0; kb < K2; kb += 32) {
        v8h b8;
#pragma unroll
        for (int e = 0; e < 8; ++e) {
            int ph = phase + esm[e >> 1];
            if (ph >= L) ph -= L;
            float fr = (float)ph * invL;         // revolutions
            float val = (e & 1) ? -hw_sinpi2(fr) : hw_cospi2(fr);
            b8[e] = (_Float16)val;
        }
        *(v8h*)&bb[lbase] = b8;
        phase += step16;
        if (phase >= L) phase -= L;
        __syncthreads();

        v16h af;
        {
            int ks = khalf * 8;
#pragma unroll
            for (int j = 0; j < 8; ++j) {
                af[j]     = (_Float16)rowPtr[kb + ks + j];
                af[8 + j] = (_Float16)rowPtr[kb + 16 + ks + j];
            }
        }
#pragma unroll
        for (int t = 0; t < 4; ++t) {
            const v16h* pb = (const v16h*)&bb[(t * 32 + lane) << 4];
            acc[t] = wmma16x16x32(af, *pb, acc[t]);
        }
        __syncthreads();
    }
#pragma unroll
    for (int v = 0; v < 8; ++v) {
        int orow = rowBase + v + 8 * khalf;
        if (orow < R) {
#pragma unroll
            for (int t = 0; t < 4; ++t) {
                int so = nb0 + t * 16 + (lane & 15);
                out[(size_t)orow * L + so] = acc[t][v];
            }
        }
    }
}

// ---------------------------------------------------------------------------
// Channel matmul (WMMA): out[b][co][s] = act(bias[co] + sum_ci inp[b][ci][s]*W)
//   transW==0: W[co*Ci+ci] (Conv1d k=1 weight [Co,Ci])
//   transW==1: W[ci*Co+co] (dense [Ci,Co], used for fc1)
// block = 128 (4 waves); wave covers 16 Co rows; N = 64 positions per block.
// ---------------------------------------------------------------------------
__global__ void __launch_bounds__(128)
chan_matmul_kernel(const float* __restrict__ inp, const float* __restrict__ W,
                   const float* __restrict__ bias, float* __restrict__ out,
                   int Ci, int Co, int L, int transW, int doGelu) {
    int b    = blockIdx.z;
    int sb0  = blockIdx.x * 64;
    int lane = threadIdx.x & 31;
    int wave = threadIdx.x >> 5;
    int coBase = blockIdx.y * 64 + wave * 16;
    int mrow   = lane & 15;
    int khalf  = lane >> 4;
    int corow  = coBase + mrow;
    if (corow >= Co) corow = Co - 1;
    const float* ib = inp + (size_t)b * Ci * L;

    v8f acc[4] = {};
    for (int cb = 0; cb < Ci; cb += 32) {
        v16h af;
        {
            int ks = khalf * 8;
#pragma unroll
            for (int j = 0; j < 8; ++j) {
                int ci0 = cb + ks + j;
                int ci1 = ci0 + 16;
                float w0 = transW ? W[(size_t)ci0 * Co + corow]
                                  : W[(size_t)corow * Ci + ci0];
                float w1 = transW ? W[(size_t)ci1 * Co + corow]
                                  : W[(size_t)corow * Ci + ci1];
                af[j]     = (_Float16)w0;
                af[8 + j] = (_Float16)w1;
            }
        }
#pragma unroll
        for (int t = 0; t < 4; ++t) {
            v16h bf;
            int s = sb0 + t * 16 + (lane & 15);
#pragma unroll
            for (int j = 0; j < 16; ++j) {
                int ci = cb + 8 * khalf + (j & 7) + 16 * (j >> 3);
                bf[j] = (_Float16)ib[(size_t)ci * L + s];
            }
            acc[t] = wmma16x16x32(af, bf, acc[t]);
        }
    }
#pragma unroll
    for (int v = 0; v < 8; ++v) {
        int co = coBase + v + 8 * khalf;
        if (co < Co) {
            float bz = bias ? bias[co] : 0.0f;
#pragma unroll
            for (int t = 0; t < 4; ++t) {
                int s = sb0 + t * 16 + (lane & 15);
                float r = acc[t][v] + bz;
                if (doGelu) r = gelu_exact(r);
                out[((size_t)b * Co + co) * L + s] = r;
            }
        }
    }
}

// ---------------------------------------------------------------------------
// combine: dest[b][chanOff+co][s'] = gelu(a[b][co][s'] + bias[co] + lerp(z))
// (align_corners=True linear interp of the pointwise-conv branch)
// ---------------------------------------------------------------------------
__global__ void __launch_bounds__(256)
combine_kernel(const float* __restrict__ aB, const float* __restrict__ zB,
               const float* __restrict__ bias, float* __restrict__ dest,
               int Bn, int Co, int Lin, int Lout, int CoTot, int chanOff) {
    size_t idx = (size_t)blockIdx.x * 256 + threadIdx.x;
    size_t total = (size_t)Bn * Co * Lout;
    if (idx >= total) return;
    int s = (int)(idx % Lout);
    size_t r = idx / Lout;
    int co = (int)(r % Co);
    int b  = (int)(r / Co);
    float pos = (float)s * ((float)(Lin - 1) / (float)(Lout - 1));
    int lo = (int)floorf(pos);
    int hi = lo + 1; if (hi > Lin - 1) hi = Lin - 1;
    float f = pos - (float)lo;
    const float* zr = zB + ((size_t)b * Co + co) * Lin;
    float z = zr[lo] * (1.0f - f) + zr[hi] * f;
    float val = gelu_exact(aB[idx] + z + bias[co]);
    dest[((size_t)b * CoTot + chanOff + co) * Lout + s] = val;
}

// ---------------------------------------------------------------------------
// channel-range concat copy: dest[b][chanOff+c][s] = src[b][c][s]
// ---------------------------------------------------------------------------
__global__ void __launch_bounds__(256)
concat_copy_kernel(const float* __restrict__ src, float* __restrict__ dest,
                   int Bn, int C, int L, int CoTot, int chanOff) {
    size_t idx = (size_t)blockIdx.x * 256 + threadIdx.x;
    size_t total = (size_t)Bn * C * L;
    if (idx >= total) return;
    int s = (int)(idx % L);
    size_t r = idx / L;
    int c = (int)(r % C);
    int b = (int)(r / C);
    dest[((size_t)b * CoTot + chanOff + c) * L + s] = src[idx];
}

// ---------------------------------------------------------------------------
// fc2: y6[b][o][s] = b2[o] + sum_j h1[b][j][s]*W2[j][o]   (o<6)
// ---------------------------------------------------------------------------
__global__ void __launch_bounds__(256)
fc2_kernel(const float* __restrict__ h1, const float* __restrict__ W2,
           const float* __restrict__ b2, float* __restrict__ y6) {
    __shared__ float sW[256 * 6];
    int tid = threadIdx.x;
    for (int i = tid; i < 256 * 6; i += 256) sW[i] = W2[i];
    __syncthreads();
    int idx = blockIdx.x * 256 + tid;   // over B*6400
    int s = idx % 6400;
    int b = idx / 6400;
    float acc[6];
#pragma unroll
    for (int o = 0; o < 6; ++o) acc[o] = b2[o];
    const float* hp = h1 + (size_t)b * 256 * 6400 + s;
    for (int j = 0; j < 256; ++j) {
        float v = hp[(size_t)j * 6400];
#pragma unroll
        for (int o = 0; o < 6; ++o) acc[o] += v * sW[j * 6 + o];
    }
#pragma unroll
    for (int o = 0; o < 6; ++o)
        y6[((size_t)b * 6 + o) * 6400 + s] = acc[o];
}

// tail mean over s of channels 3..5 -> meanB[b*3 + (o-3)]
__global__ void __launch_bounds__(256)
tail_mean_kernel(const float* __restrict__ y6, float* __restrict__ meanB) {
    __shared__ float red[256];
    int blk = blockIdx.x;        // 0..95
    int b = blk / 3;
    int o = 3 + blk % 3;
    int tid = threadIdx.x;
    float s = 0.0f;
    const float* row = y6 + ((size_t)b * 6 + o) * 6400;
    for (int i = tid; i < 6400; i += 256) s += row[i];
    red[tid] = s;
    __syncthreads();
    for (int st = 128; st > 0; st >>= 1) {
        if (tid < st) red[tid] += red[tid + st];
        __syncthreads();
    }
    if (tid == 0) meanB[blk] = red[0] * (1.0f / 6400.0f);
}

// final: out[b][o][s<6000] = tanh(o<3 ? y6[b][o][s] : mean[b][o-3])
__global__ void __launch_bounds__(256)
final_kernel(const float* __restrict__ y6, const float* __restrict__ meanB,
             float* __restrict__ out) {
    size_t idx = (size_t)blockIdx.x * 256 + threadIdx.x;
    if (idx >= (size_t)32 * 6 * 6000) return;
    int s = (int)(idx % 6000);
    size_t r = idx / 6000;
    int o = (int)(r % 6);
    int b = (int)(r / 6);
    float v = (o < 3) ? y6[((size_t)b * 6 + o) * 6400 + s]
                      : meanB[b * 3 + (o - 3)];
    out[idx] = tanhf(v);
}

// ---------------------------------------------------------------------------
// Host orchestration
// ---------------------------------------------------------------------------
static inline unsigned cdiv(long long a, long long b) {
    return (unsigned)((a + b - 1) / b);
}

extern "C" void kernel_launch(void* const* d_in, const int* in_sizes, int n_in,
                              void* d_out, int out_size, void* d_ws, size_t ws_size,
                              hipStream_t stream) {
    (void)in_sizes; (void)n_in; (void)out_size; (void)ws_size;
    const int B = 32;

    const float* x      = (const float*)d_in[0];   // [32,6400,8]
    const float* label  = (const float*)d_in[1];   // [32,1,4]
    const float* fc0_w  = (const float*)d_in[2];   // [12,64]
    const float* fc0_b  = (const float*)d_in[3];   // [64]
    const float* scw[8], *pww[8], *pwb[8];
    for (int i = 0; i < 8; ++i) {
        scw[i] = (const float*)d_in[4 + 3 * i];    // complex64 [ci,co,m]
        pww[i] = (const float*)d_in[5 + 3 * i];    // [co,ci]
        pwb[i] = (const float*)d_in[6 + 3 * i];    // [co]
    }
    const float* fc1_w = (const float*)d_in[28];   // [128,256]
    const float* fc1_b = (const float*)d_in[29];   // [256]
    const float* fc2_w = (const float*)d_in[30];   // [256,6]
    const float* fc2_b = (const float*)d_in[31];   // [6]

    // ---- workspace layout (floats) with reuse: c4 aliases c29, c5 aliases c3
    float* ws   = (float*)d_ws;
    float* h    = ws;                                   // 32*64*6400
    float* c0   = h    + (size_t)32 * 64  * 6400;       // 32*96*4800
    float* c1   = c0   + (size_t)32 * 96  * 4800;       // 32*192*3200
    float* c2   = c1   + (size_t)32 * 192 * 3200;       // 32*384*1600
    float* c21  = c2   + (size_t)32 * 384 * 1600;       // 32*768*800
    float* c29  = c21  + (size_t)32 * 768 * 800;        // 39.32M (also c4)
    float* c3   = c29  + (size_t)39321600;              // 39.32M (also c5)
    float* specA= c3   + (size_t)39321600;              // 39.32M
    float* specB= specA+ (size_t)39321600;              // 19.66M
    float* aBuf = specB+ (size_t)19660800;              // 19.66M
    float* zBuf = aBuf + (size_t)19660800;              // 39.32M
    float* h1   = zBuf + (size_t)39321600;              // 32*256*6400
    float* y6   = h1   + (size_t)32 * 256 * 6400;       // 32*6*6400
    float* meanB= y6   + (size_t)32 * 6 * 6400;         // 96
    float* c4 = c29;
    float* c5 = c3;

    // ---- stage configuration
    const int Ci_[8] = {64, 96, 192, 384, 768, 768, 384, 192};
    const int Co_[8] = {96, 192, 384, 768, 384, 192, 96, 64};
    const int Lo_[8] = {4800, 3200, 1600, 800, 1600, 3200, 4800, 6400};
    const int Md_[8] = {2400, 1600, 800, 400, 400, 800, 1600, 2400};
    const int Li_[8] = {6400, 4800, 3200, 1600, 800, 1600, 3200, 4800};
    float* ins_[8]  = {h,  c0, c1, c2,  c21, c29, c3, c4};
    float* dst_[8]  = {c0, c1, c2, c21, c29, c3,  c4, c5};
    const int CoT_[8] = {96, 192, 384, 768, 768, 384, 192, 128};

    // ---- fc0
    fc0_kernel<<<dim3((B * 6400) / 256), dim3(256), 0, stream>>>(
        x, label, fc0_w, fc0_b, h);

    // ---- 8 spectral/pointwise blocks
    for (int i = 0; i < 8; ++i) {
        const int Ci = Ci_[i], Co = Co_[i], Lout = Lo_[i], m = Md_[i], Lin = Li_[i];
        const int Rin = B * Ci, Rout = B * Co;

        dft_fwd_kernel<<<dim3(cdiv(m, 64), Rin / 128), dim3(256), 0, stream>>>(
            ins_[i], specA, Rin, Lin, m);

        spec_einsum_kernel<<<dim3(m, cdiv(Co, 64)), dim3(256), 0, stream>>>(
            specA, scw[i], specB, Ci, Co, m, 1.0f / (float)Lout);

        dft_inv_kernel<<<dim3(Lout / 64, Rout / 128), dim3(256), 0, stream>>>(
            specB, aBuf, Rout, 2 * m, Lout);

        chan_matmul_kernel<<<dim3(Lin / 64, cdiv(Co, 64), B), dim3(128), 0, stream>>>(
            ins_[i], pww[i], nullptr, zBuf, Ci, Co, Lin, /*transW=*/0, /*gelu=*/0);

        combine_kernel<<<dim3(cdiv((long long)B * Co * Lout, 256)), dim3(256), 0, stream>>>(
            aBuf, zBuf, pwb[i], dst_[i], B, Co, Lin, Lout, CoT_[i], 0);

        // U-Net skip concats
        if (i == 4)      // c29 = cat(block4, c2)
            concat_copy_kernel<<<dim3(cdiv((long long)B * 384 * 1600, 256)), dim3(256), 0, stream>>>(
                c2, c29, B, 384, 1600, 768, 384);
        else if (i == 5) // c3 = cat(block5, c1)
            concat_copy_kernel<<<dim3(cdiv((long long)B * 192 * 3200, 256)), dim3(256), 0, stream>>>(
                c1, c3, B, 192, 3200, 384, 192);
        else if (i == 6) // c4 = cat(block6, c0)
            concat_copy_kernel<<<dim3(cdiv((long long)B * 96 * 4800, 256)), dim3(256), 0, stream>>>(
                c0, c4, B, 96, 4800, 192, 96);
        else if (i == 7) // c5 = cat(block7, h)
            concat_copy_kernel<<<dim3(cdiv((long long)B * 64 * 6400, 256)), dim3(256), 0, stream>>>(
                h, c5, B, 64, 6400, 128, 64);
    }

    // ---- fc1 (GEMM over channels, exact GELU), then fc2 + tail + tanh
    chan_matmul_kernel<<<dim3(6400 / 64, cdiv(256, 64), B), dim3(128), 0, stream>>>(
        c5, fc1_w, fc1_b, h1, 128, 256, 6400, /*transW=*/1, /*gelu=*/1);

    fc2_kernel<<<dim3((B * 6400) / 256), dim3(256), 0, stream>>>(
        h1, fc2_w, fc2_b, y6);

    tail_mean_kernel<<<dim3(B * 3), dim3(256), 0, stream>>>(y6, meanB);

    final_kernel<<<dim3(cdiv((long long)B * 6 * 6000, 256)), dim3(256), 0, stream>>>(
        y6, meanB, (float*)d_out);
}
